// AnomalyAttention_45234595561776
// MI455X (gfx1250) — matrix operand order repaired and checked by reading, not compiled, not run on hardware
//
#include <hip/hip_runtime.h>
#include <hip/hip_bf16.h>

// CDNA5 / gfx1250 bf16 WMMA attention forward (AnomalyAttention).
// B=4, L=2048, H=8, E=D=64.  d_out = [V (B,L,H,D) | series (B,H,L,L)].

typedef __attribute__((ext_vector_type(16))) __bf16 v16bf;
typedef __attribute__((ext_vector_type(8)))  float  v8f;
typedef __attribute__((ext_vector_type(4)))  float  v4f;

#define B_  4
#define L_  2048
#define H_  8
#define E_  64
#define D_  64
#define LOG2E 1.44269504088896f

__global__ __launch_bounds__(32)
void anomaly_attn_fwd(const float* __restrict__ Q,
                      const float* __restrict__ K,
                      const float* __restrict__ V,
                      float* __restrict__ Vout,
                      float* __restrict__ series)
{
    // P tile (16 rows x full key length) as bf16: 64 KB (<= 320 KB/WGP on CDNA5)
    __shared__ __bf16 Pbuf[16 * L_];
    // Transposed bf16 V chunk: Vt[d][kk], 64 x 32 -> 4 KB
    __shared__ __bf16 Vt[D_ * 32];
    __shared__ float  rinv_s[16];

    const int lane = threadIdx.x;      // 0..31 (wave32)
    const int n    = lane & 15;        // column / M index depending on role
    const int half = lane >> 4;        // 0 or 1

    const int tile = blockIdx.x & (L_ / 16 - 1);   // 128 row tiles
    const int bh   = blockIdx.x >> 7;
    const int h    = bh & (H_ - 1);
    const int b    = bh >> 3;

    const int m0     = tile * 16;
    const int nkeys  = m0 + 16;        // causal extent for this tile
    const int jtiles = tile + 1;

    // ------------------------------------------------------------------
    // Load Q tile (16 x 64) into WMMA-A layout, two K=32 chunks.
    // A layout: lane -> row M = lane&15; elements 0..7 -> K = half*8+e,
    //           elements 8..15 -> K = 16 + half*8 + (e-8).
    // ------------------------------------------------------------------
    v16bf aQ[2];
    {
        const float* qrow = Q + (((size_t)b * L_ + (m0 + n)) * H_ + h) * E_;
        #pragma unroll
        for (int c = 0; c < 2; ++c) {
            const float* p0 = qrow + c * 32 + half * 8;
            const float* p1 = qrow + c * 32 + 16 + half * 8;
            #pragma unroll
            for (int e = 0; e < 8; ++e) {
                aQ[c][e]     = (__bf16)p0[e];
                aQ[c][8 + e] = (__bf16)p1[e];
            }
        }
    }

    // ------------------------------------------------------------------
    // Pass 1: scores = Q K^T (bf16 WMMA), exp (raw v_exp_f32), P -> LDS,
    // row sums.  Bulk tiles need no causal mask; only the diagonal does.
    // ------------------------------------------------------------------
    float rsum[8];
    #pragma unroll
    for (int v = 0; v < 8; ++v) rsum[v] = 0.0f;

    const float scl = 0.125f * LOG2E;  // 1/sqrt(E) folded into log2 domain
    const __bf16* kbase_ptr = K + (((size_t)b * L_ + n) * H_ + h) * E_ ? nullptr : nullptr; // (unused)

    // Bulk (fully unmasked) key tiles: j in [0, tile)
    for (int j = 0; j < tile; ++j) {
        const float* krow = K + (((size_t)b * L_ + (j * 16 + n)) * H_ + h) * E_;
        v8f acc = {};
        #pragma unroll
        for (int c = 0; c < 2; ++c) {
            v16bf bK;
            const float* p = krow + c * 32 + half * 8;
            #pragma unroll
            for (int e = 0; e < 16; ++e) bK[e] = (__bf16)p[e < 8 ? e : e + 8];
            acc = __builtin_amdgcn_wmma_f32_16x16x32_bf16(
                      false, aQ[c], false, bK, (short)0, acc, false, false);
        }
        // C/D layout: VGPR v -> row (v + 8*half), col n.
        __bf16* prow = &Pbuf[half * 8 * L_ + j * 16 + n];
        #pragma unroll
        for (int v = 0; v < 8; ++v) {
            const float p = __builtin_amdgcn_exp2f(acc[v] * scl);
            rsum[v] += p;
            prow[v * L_] = (__bf16)p;
        }
    }

    // Diagonal key tile (j == tile): branch-free causal mask.
    {
        const float* krow = K + (((size_t)b * L_ + (m0 + n)) * H_ + h) * E_;
        v8f acc = {};
        #pragma unroll
        for (int c = 0; c < 2; ++c) {
            v16bf bK;
            const float* p = krow + c * 32 + half * 8;
            #pragma unroll
            for (int e = 0; e < 16; ++e) bK[e] = (__bf16)p[e < 8 ? e : e + 8];
            acc = __builtin_amdgcn_wmma_f32_16x16x32_bf16(
                      false, aQ[c], false, bK, (short)0, acc, false, false);
        }
        __bf16* prow = &Pbuf[half * 8 * L_ + m0 + n];
        #pragma unroll
        for (int v = 0; v < 8; ++v) {
            const int   row = v + half * 8;
            const float e   = __builtin_amdgcn_exp2f(acc[v] * scl);
            const float p   = (n <= row) ? e : 0.0f;   // v_cndmask, no branch
            rsum[v] += p;
            prow[v * L_] = (__bf16)p;
        }
    }

    // Pad one extra 16-key block with zeros if the causal extent is an odd
    // multiple of 16, so K=32 PV chunks read valid zeros.
    if (jtiles & 1) {
        #pragma unroll
        for (int v = 0; v < 8; ++v) {
            const int row = v + half * 8;
            Pbuf[row * L_ + nkeys + n] = (__bf16)0.0f;
        }
    }

    // Reduce row sums across the 16 lanes of each half (column dimension).
    #pragma unroll
    for (int v = 0; v < 8; ++v) {
        float r = rsum[v];
        r += __shfl_xor(r, 1, 32);
        r += __shfl_xor(r, 2, 32);
        r += __shfl_xor(r, 4, 32);
        r += __shfl_xor(r, 8, 32);
        rsum[v] = r;
    }
    if (n == 0) {
        #pragma unroll
        for (int v = 0; v < 8; ++v)
            rinv_s[v + half * 8] = 1.0f / rsum[v];
    }
    __syncthreads();

    // ------------------------------------------------------------------
    // Pass 2: O = P V (bf16 WMMA over 32-key chunks), normalize at end.
    // ------------------------------------------------------------------
    v8f o[4];
    #pragma unroll
    for (int nt = 0; nt < 4; ++nt) o[nt] = (v8f){};

    const int kchunks = (nkeys + 31) >> 5;
    for (int kc = 0; kc < kchunks; ++kc) {
        const int kbase = kc * 32;

        // Stage V chunk (32 keys x 64 dims): float4 coalesced loads,
        // transposed bf16 stores into LDS.
        {
            const float* vbase = V + (((size_t)b * L_ + kbase) * H_ + h) * D_;
            #pragma unroll
            for (int it = 0; it < 16; ++it) {
                const int idx = it * 128 + lane * 4;     // element in 32x64 chunk
                const int kk  = idx >> 6;
                const int d   = idx & 63;
                // V rows are H*D apart; within a row d is contiguous.
                const v4f vv = *(const v4f*)(vbase + (size_t)kk * (H_ * D_) + d);
                #pragma unroll
                for (int q = 0; q < 4; ++q)
                    Vt[(d + q) * 32 + kk] = (__bf16)vv[q];
            }
        }
        __syncthreads();

        // A = P chunk (16 x 32), rows from LDS (contiguous 8+8 bf16).
        v16bf aP;
        {
            const __bf16* pr = &Pbuf[n * L_ + kbase + half * 8];
            #pragma unroll
            for (int e = 0; e < 8; ++e) {
                aP[e]     = pr[e];
                aP[8 + e] = pr[16 + e];
            }
        }

        // B = V chunk (32 x 16 per n-tile): element e<8 -> k = half*8 + e, e>=8 -> k = 16 + half*8 + (e-8).
        #pragma unroll
        for (int nt = 0; nt < 4; ++nt) {
            v16bf bV;
            const __bf16* p = &Vt[(nt * 16 + n) * 32 + half * 8];
            #pragma unroll
            for (int e = 0; e < 16; ++e) bV[e] = p[e < 8 ? e : e + 8];
            o[nt] = __builtin_amdgcn_wmma_f32_16x16x32_bf16(
                        false, aP, false, bV, (short)0, o[nt], false, false);
        }
        __syncthreads();
    }

    // Write V output: row (v + 8*half), dim nt*16 + n, normalized.
    #pragma unroll
    for (int nt = 0; nt < 4; ++nt) {
        #pragma unroll
        for (int v = 0; v < 8; ++v) {
            const int row = v + half * 8;
            const int d   = nt * 16 + n;
            Vout[(((size_t)b * L_ + (m0 + row)) * H_ + h) * D_ + d] =
                o[nt][v] * rinv_s[row];
        }
    }

    // ------------------------------------------------------------------
    // Write series rows (normalized P, zeros above diagonal), float4 stores.
    // ------------------------------------------------------------------
    float* srow_base = series + (((size_t)b * H_ + h) * L_ + m0) * (size_t)L_;
    for (int row = 0; row < 16; ++row) {
        float*      out = srow_base + (size_t)row * L_;
        const float ri  = rinv_s[row];
        const int   lim = m0 + row;
        for (int c0 = lane * 4; c0 < L_; c0 += 128) {
            v4f w;
            #pragma unroll
            for (int q = 0; q < 4; ++q) {
                const int c = c0 + q;
                w[q] = (c <= lim) ? (float)Pbuf[row * L_ + c] * ri : 0.0f;
            }
            *(v4f*)(out + c0) = w;
        }
    }
}

extern "C" void kernel_launch(void* const* d_in, const int* in_sizes, int n_in,
                              void* d_out, int out_size, void* d_ws, size_t ws_size,
                              hipStream_t stream) {
    (void)in_sizes; (void)n_in; (void)d_ws; (void)ws_size; (void)out_size;
    const float* Q = (const float*)d_in[0];
    const float* K = (const float*)d_in[1];
    const float* V = (const float*)d_in[2];
    // d_in[3] = sigma, unused in forward (matches reference)

    float* Vout   = (float*)d_out;
    float* series = (float*)d_out + (size_t)B_ * L_ * H_ * D_;

    const int blocks = B_ * H_ * (L_ / 16);   // 4096 row tiles
    anomaly_attn_fwd<<<blocks, 32, 0, stream>>>(Q, K, V, Vout, series);
}
